// CrossAttention_44959717654835
// MI455X (gfx1250) — matrix operand order repaired and checked by reading, not compiled, not run on hardware
//
#include <hip/hip_runtime.h>
#include <hip/hip_bf16.h>

// ---------------------------------------------------------------------------
// CrossAttention for MI455X (gfx1250), bf16 WMMA pipeline.
//   B=4, S=L=2048, E=512, CE=768, H=8, D=64
// d_out = [ out f32 (B*S*E) | probs f32 (B*H*S*L) ]
// Roofline: probs write (537MB -> ~23us @23.3TB/s) ~ balanced with ~56GF of
// WMMA-bf16 math, so: touch probs exactly once, do all GEMMs with
// v_wmma_f32_16x16x32_bf16, keep full 16x2048 score strips in the 320KB LDS.
// ---------------------------------------------------------------------------

typedef unsigned short u16;
typedef unsigned int   u32;
typedef unsigned long long u64;
typedef __bf16 bf16_t;
typedef bf16_t v16bf __attribute__((ext_vector_type(16)));
typedef float  v8f   __attribute__((ext_vector_type(8)));

#define CA_B 4
#define CA_S 2048
#define CA_L 2048
#define CA_E 512
#define CA_CE 768
#define CA_H 8
#define CA_D 64

__device__ __forceinline__ u16 f2bf(float f) {
    union { float f; u32 u; } x; x.f = f;
    u32 u = x.u;
    u32 r = (u + 0x7FFFu + ((u >> 16) & 1u)) >> 16;  // round-to-nearest-even
    return (u16)r;
}
__device__ __forceinline__ bf16_t us2bf(u16 u) {
    union { u16 u; bf16_t b; } c; c.u = u; return c.b;
}
__device__ __forceinline__ v8f wmma_bf16(v16bf a, v16bf b, v8f c) {
    // D = A(16x32 bf16) * B(32x16 bf16) + C(16x16 f32)
    return __builtin_amdgcn_wmma_f32_16x16x32_bf16(
        /*neg_a=*/false, a, /*neg_b=*/false, b,
        /*c_mod=*/(short)0, c, /*reuse_a=*/false, /*reuse_b=*/false);
}

// ---------------------------------------------------------------------------
// f32 -> bf16 cast, vectorized: float4 in (16B), 4x bf16 packed u64 out (8B).
// All array sizes here are multiples of 4.
// ---------------------------------------------------------------------------
__global__ void cast_f32_bf16_kernel(const float* __restrict__ in,
                                     u16* __restrict__ out, long n4) {
    long i = (long)blockIdx.x * blockDim.x + threadIdx.x;
    long stride = (long)gridDim.x * blockDim.x;
    for (; i < n4; i += stride) {
        float4 f = *(const float4*)(in + i * 4);
        union { u16 u[4]; u64 q; } p;
        p.u[0] = f2bf(f.x); p.u[1] = f2bf(f.y);
        p.u[2] = f2bf(f.z); p.u[3] = f2bf(f.w);
        *(u64*)(out + i * 4) = p.q;
    }
}

// ---------------------------------------------------------------------------
// C[M,N] = A[M,K](bf16) * W[K,N](bf16) + bias; output bf16 or f32.
// Block: 128 threads = 4 waves; tile 32(M) x 64(N); each wave owns 16 N cols
// and both 16-row M subtiles (B-fragment reused across the two WMMAs).
// LDS staging via 16B uint4 loads.
// ---------------------------------------------------------------------------
__global__ __launch_bounds__(128)
void gemm_wmma_kernel(const u16* __restrict__ A, const u16* __restrict__ Bm,
                      const float* __restrict__ bias,
                      u16* __restrict__ Cb, float* __restrict__ Cf,
                      int M, int N, int K) {
    __shared__ __align__(16) u16 As[32][32];   // 2 KB
    __shared__ __align__(16) u16 Bs[32][64];   // 4 KB

    const int tid  = threadIdx.x;
    const int lane = tid & 31;
    const int w    = tid >> 5;
    const int m0   = blockIdx.y * 32;
    const int n0   = blockIdx.x * 64;
    const int m    = lane & 15;
    const int hi8  = (lane >> 4) * 8;

    const u16* Abase = A + (size_t)m0 * K;        // advance by +32 per k-step
    const u16* Bbase = Bm + n0;                   // advance by +32*N per k-step

    v8f c0 = {0.f,0.f,0.f,0.f,0.f,0.f,0.f,0.f};
    v8f c1 = {0.f,0.f,0.f,0.f,0.f,0.f,0.f,0.f};

    // staging indices (hoisted)
    const int ar = tid >> 2, ac = (tid & 3) * 8;              // A: 128 x uint4
    const int br0 = tid >> 3, bc0 = (tid & 7) * 8;            // B: 256 x uint4
    const int br1 = (tid + 128) >> 3, bc1 = ((tid + 128) & 7) * 8;

    for (int k0 = 0; k0 < K; k0 += 32) {
        __syncthreads();
        // A tile [32][32]: one 16B load per thread
        *(uint4*)&As[ar][ac] = *(const uint4*)(Abase + (size_t)ar * K + k0 + ac);
        // B tile [32][64]: two 16B loads per thread
        *(uint4*)&Bs[br0][bc0] =
            *(const uint4*)(Bbase + (size_t)(k0 + br0) * N + bc0);
        *(uint4*)&Bs[br1][bc1] =
            *(const uint4*)(Bbase + (size_t)(k0 + br1) * N + bc1);
        __syncthreads();

        v16bf a0, a1, b;
        #pragma unroll
        for (int v = 0; v < 8; ++v) {
            // A 16x32 layout: lane-halves split K {0..15 | 16..31} w/ 8-offset
            int kb = (v < 4 ? 2 * v : 16 + 2 * (v - 4)) + hi8;
            a0[2*v]   = us2bf(As[m][kb]);
            a0[2*v+1] = us2bf(As[m][kb + 1]);
            a1[2*v]   = us2bf(As[16 + m][kb]);
            a1[2*v+1] = us2bf(As[16 + m][kb + 1]);
            // B 32x16 layout: lane = N, K same order as A
            b[2*v]   = us2bf(Bs[kb][w * 16 + m]);
            b[2*v+1] = us2bf(Bs[kb + 1][w * 16 + m]);
        }
        c0 = wmma_bf16(a0, b, c0);
        c1 = wmma_bf16(a1, b, c1);
    }

    const int n = lane & 15;
    const int col = n0 + w * 16 + n;
    const float bb = bias[col];
    #pragma unroll
    for (int r = 0; r < 8; ++r) {
        int row0 = m0 + r + hi8;          // C/D: M = vgpr + 8*(lane>=16)
        int row1 = row0 + 16;
        float v0 = c0[r] + bb;
        float v1 = c1[r] + bb;
        if (Cf) {
            Cf[(size_t)row0 * N + col] = v0;
            Cf[(size_t)row1 * N + col] = v1;
        } else {
            Cb[(size_t)row0 * N + col] = f2bf(v0);
            Cb[(size_t)row1 * N + col] = f2bf(v1);
        }
    }
}

// ---------------------------------------------------------------------------
// Fused attention: per block = 16 query rows of one (b,h).
//  scores(16x2048, f32 in LDS) -> softmax -> probs once to HBM -> PV (WMMA).
// 4 waves: scores phase splits L (512 cols each); PV phase splits D (16 each).
// LDS ~141KB -> 2 blocks per WGP.
// ---------------------------------------------------------------------------
__global__ __launch_bounds__(128)
void attn_fused_kernel(const u16* __restrict__ qbf, const u16* __restrict__ kbf,
                       const u16* __restrict__ vbf, u16* __restrict__ ctxbf,
                       float* __restrict__ probs) {
    __shared__ float sc[16 * CA_L];                       // 128 KB scores strip
    __shared__ __align__(16) u16 kst[4][16 * 64];         // 8 KB  k staging/wave
    __shared__ __align__(16) u16 vst[32 * 64];            // 4 KB  v staging
    __shared__ float red[16][8];
    __shared__ float rowmax[16], rowsum[16];

    const int tid  = threadIdx.x;
    const int lane = tid & 31;
    const int w    = tid >> 5;

    const int strip   = blockIdx.x;            // B*H*(S/16) = 4096
    const int s_strip = strip & (CA_S / 16 - 1);
    const int bh      = strip >> 7;            // S/16 == 128
    const int h       = bh & (CA_H - 1);
    const int b       = bh >> 3;
    const int s0      = s_strip * 16;

    const int m   = lane & 15;
    const int hi8 = (lane >> 4) * 8;

    // --- q fragments: A-matrix, 2 k-steps over D=64 ---
    v16bf aq[2];
    {
        const u16* qrow = qbf + (size_t)(b * CA_S + s0 + m) * CA_E + h * CA_D;
        #pragma unroll
        for (int ks = 0; ks < 2; ++ks)
            #pragma unroll
            for (int v = 0; v < 8; ++v) {
                int kb = (v < 4 ? 2 * v : 16 + 2 * (v - 4)) + hi8;
                aq[ks][2*v]   = us2bf(qrow[ks * 32 + kb]);
                aq[ks][2*v+1] = us2bf(qrow[ks * 32 + kb + 1]);
            }
    }

    // --- scores: wave w covers L cols [w*512, w*512+512), tiles of 16 ---
    const int lbase = w * (CA_L / 4);
    for (int lt = 0; lt < (CA_L / 4) / 16; ++lt) {
        const int l0 = lbase + lt * 16;
        {   // stage k tile [16 l][64 d] via 16B vector loads (per-wave, no barrier)
            uint4* dst = (uint4*)&kst[w][0];
            const u16* kb0 = kbf + (size_t)(b * CA_L + l0) * CA_E + h * CA_D;
            for (int i = lane; i < 128; i += 32) {
                int r = i >> 3, cc = (i & 7) * 8;
                const uint4* src = (const uint4*)(kb0 + (size_t)r * CA_E + cc);
                __builtin_prefetch(src + 512, 0, 1);   // hint 8 rows (half a tile) ahead
                dst[i] = *src;
            }
        }
        v16bf b0, b1;
        #pragma unroll
        for (int v = 0; v < 8; ++v) {
            int kb = (v < 4 ? 2 * v : 16 + 2 * (v - 4)) + hi8;
            b0[2*v]   = us2bf(kst[w][m * 64 + kb]);        // Kᵀ: (d, l)
            b0[2*v+1] = us2bf(kst[w][m * 64 + kb + 1]);
            b1[2*v]   = us2bf(kst[w][m * 64 + 32 + kb]);
            b1[2*v+1] = us2bf(kst[w][m * 64 + 32 + kb + 1]);
        }
        v8f c = {0.f,0.f,0.f,0.f,0.f,0.f,0.f,0.f};
        c = wmma_bf16(aq[0], b0, c);
        c = wmma_bf16(aq[1], b1, c);
        #pragma unroll
        for (int r = 0; r < 8; ++r)
            sc[(r + hi8) * CA_L + l0 + (lane & 15)] = c[r] * 0.125f; // 1/sqrt(64)
    }
    __syncthreads();

    // --- row softmax (8 threads per row), normalized probs -> HBM once ---
    {
        const int row = tid >> 3;
        const int sub = tid & 7;
        float mx = -3.4e38f;
        for (int i = sub; i < CA_L; i += 8)
            mx = fmaxf(mx, sc[row * CA_L + i]);
        red[row][sub] = mx;
        __syncthreads();
        if (sub == 0) {
            float m2 = red[row][0];
            #pragma unroll
            for (int j = 1; j < 8; ++j) m2 = fmaxf(m2, red[row][j]);
            rowmax[row] = m2;
        }
        __syncthreads();
        const float rm = rowmax[row];
        float sum = 0.f;
        for (int i = sub; i < CA_L; i += 8) {
            float e = __expf(sc[row * CA_L + i] - rm);
            sc[row * CA_L + i] = e;
            sum += e;
        }
        red[row][sub] = sum;
        __syncthreads();
        if (sub == 0) {
            float s2 = 0.f;
            #pragma unroll
            for (int j = 0; j < 8; ++j) s2 += red[row][j];
            rowsum[row] = s2;
        }
        __syncthreads();
        const float inv = 1.0f / rowsum[row];
        float* prow = probs +
            (size_t)((b * CA_H + h) * CA_S + s0 + row) * CA_L;
        for (int i = sub; i < CA_L; i += 8) {
            float p = sc[row * CA_L + i] * inv;
            prow[i] = p;                 // the one-and-only probs HBM write
            sc[row * CA_L + i] = p;      // keep normalized probs for PV
        }
    }
    __syncthreads();

    // --- PV: ctx(16x64) = probs(16x2048) @ v(2048x64); wave w owns 16 D cols ---
    v8f c2 = {0.f,0.f,0.f,0.f,0.f,0.f,0.f,0.f};
    for (int ks = 0; ks < CA_L / 32; ++ks) {
        const int l0 = ks * 32;
        __syncthreads();
        {   // stage v tile [32 l][64 d] cooperatively (256 x 16B loads)
            uint4* dst = (uint4*)vst;
            const u16* vb0 = vbf + (size_t)(b * CA_L + l0) * CA_E + h * CA_D;
            #pragma unroll
            for (int i = tid; i < 256; i += 128) {
                int r = i >> 3, cc = (i & 7) * 8;
                dst[i] = *(const uint4*)(vb0 + (size_t)r * CA_E + cc);
            }
        }
        __syncthreads();
        v16bf ap, bv;
        #pragma unroll
        for (int v = 0; v < 8; ++v) {
            int kb = (v < 4 ? 2 * v : 16 + 2 * (v - 4)) + hi8;
            ap[2*v]   = us2bf(f2bf(sc[m * CA_L + l0 + kb]));     // probs -> bf16
            ap[2*v+1] = us2bf(f2bf(sc[m * CA_L + l0 + kb + 1]));
            bv[2*v]   = us2bf(vst[kb * 64 + w * 16 + m]);
            bv[2*v+1] = us2bf(vst[(kb + 1) * 64 + w * 16 + m]);
        }
        c2 = wmma_bf16(ap, bv, c2);
    }
    {
        const int n = lane & 15;
        #pragma unroll
        for (int r = 0; r < 8; ++r) {
            int mm = r + hi8;
            ctxbf[(size_t)(b * CA_S + s0 + mm) * CA_E + h * CA_D + w * 16 + n] =
                f2bf(c2[r]);
        }
    }
}

// ---------------------------------------------------------------------------
// Host launcher
// ---------------------------------------------------------------------------
extern "C" void kernel_launch(void* const* d_in, const int* in_sizes, int n_in,
                              void* d_out, int out_size, void* d_ws, size_t ws_size,
                              hipStream_t stream) {
    const float* query   = (const float*)d_in[0];
    const float* context = (const float*)d_in[1];
    const float* Wq = (const float*)d_in[2];
    const float* bq = (const float*)d_in[3];
    const float* Wk = (const float*)d_in[4];
    const float* bk = (const float*)d_in[5];
    const float* Wv = (const float*)d_in[6];
    const float* bv = (const float*)d_in[7];
    const float* Wo = (const float*)d_in[8];
    const float* bo = (const float*)d_in[9];

    float* out_f32 = (float*)d_out;                                  // B*S*E
    float* probs   = (float*)d_out + (size_t)CA_B * CA_S * CA_E;     // B*H*S*L

    // workspace layout (u16 elements), ~57 MB total
    const size_t nQ  = (size_t)CA_B * CA_S * CA_E;   // 4,194,304
    const size_t nC  = (size_t)CA_B * CA_L * CA_CE;  // 6,291,456
    u16* qbf        = (u16*)d_ws;
    u16* kbf        = qbf + nQ;
    u16* vbf        = kbf + nQ;
    u16* ctxbf      = vbf + nQ;
    u16* query_bf   = ctxbf + nQ;
    u16* context_bf = query_bf + nQ;
    u16* Wq_bf      = context_bf + nC;
    u16* Wk_bf      = Wq_bf + (size_t)CA_E * CA_E;
    u16* Wv_bf      = Wk_bf + (size_t)CA_CE * CA_E;
    u16* Wo_bf      = Wv_bf + (size_t)CA_CE * CA_E;

    // 1) casts to bf16 (vectorized: n/4 iterations)
    cast_f32_bf16_kernel<<<1024, 256, 0, stream>>>(query,   query_bf,   (long)(nQ / 4));
    cast_f32_bf16_kernel<<<1024, 256, 0, stream>>>(context, context_bf, (long)(nC / 4));
    cast_f32_bf16_kernel<<<256,  256, 0, stream>>>(Wq, Wq_bf, (long)(CA_E * CA_E / 4));
    cast_f32_bf16_kernel<<<256,  256, 0, stream>>>(Wk, Wk_bf, (long)(CA_CE * CA_E / 4));
    cast_f32_bf16_kernel<<<256,  256, 0, stream>>>(Wv, Wv_bf, (long)(CA_CE * CA_E / 4));
    cast_f32_bf16_kernel<<<256,  256, 0, stream>>>(Wo, Wo_bf, (long)(CA_E * CA_E / 4));

    const int Mq = CA_B * CA_S;   // 8192
    // 2) projections (bf16 out)
    {
        dim3 grid(CA_E / 64, Mq / 32);
        gemm_wmma_kernel<<<grid, 128, 0, stream>>>(query_bf, Wq_bf, bq,
                                                   qbf, nullptr, Mq, CA_E, CA_E);
        gemm_wmma_kernel<<<grid, 128, 0, stream>>>(context_bf, Wk_bf, bk,
                                                   kbf, nullptr, Mq, CA_E, CA_CE);
        gemm_wmma_kernel<<<grid, 128, 0, stream>>>(context_bf, Wv_bf, bv,
                                                   vbf, nullptr, Mq, CA_E, CA_CE);
    }

    // 3) fused attention (scores -> softmax -> probs -> PV)
    {
        dim3 grid(CA_B * CA_H * (CA_S / 16));   // 4096
        attn_fused_kernel<<<grid, 128, 0, stream>>>(qbf, kbf, vbf, ctxbf, probs);
    }

    // 4) output projection (f32 out)
    {
        dim3 grid(CA_E / 64, Mq / 32);
        gemm_wmma_kernel<<<grid, 128, 0, stream>>>(ctxbf, Wo_bf, bo,
                                                   nullptr, out_f32, Mq, CA_E, CA_E);
    }
}
